// AttentionSHA_80917183856888
// MI455X (gfx1250) — matrix operand order repaired and checked by reading, not compiled, not run on hardware
//
#include <hip/hip_runtime.h>

#define DEVI __device__ __forceinline__

typedef __attribute__((ext_vector_type(16))) __bf16        v16bf;
typedef __attribute__((ext_vector_type(8)))  float         v8f;
typedef __attribute__((ext_vector_type(8)))  unsigned int  v8u;

// ---------------------------------------------------------------- helpers
DEVI unsigned short f2bf(float f) {
    unsigned int x = __builtin_bit_cast(unsigned int, f);
    x += 0x7FFFu + ((x >> 16) & 1u);      // round-to-nearest-even
    return (unsigned short)(x >> 16);
}
DEVI unsigned int pack2bf(float lo, float hi) {
    return (unsigned int)f2bf(lo) | ((unsigned int)f2bf(hi) << 16);
}
DEVI v16bf as_bf(v8u u) { return __builtin_bit_cast(v16bf, u); }
DEVI v8f vzero8() {
    v8f v;
#pragma unroll
    for (int i = 0; i < 8; i++) v[i] = 0.0f;
    return v;
}

// CDNA5 async global->LDS copy (16B per lane), tracked by ASYNCcnt.
// LDS byte address = low 32 bits of the flat shared-aperture address
// (ISA 10.2).
DEVI void async_ld16(const unsigned short* lds_dst, const unsigned short* gsrc) {
    unsigned lds_off = (unsigned)(size_t)lds_dst;
    asm volatile("global_load_async_to_lds_b128 %0, %1, off"
                 :: "v"(lds_off), "v"(gsrc) : "memory");
}
DEVI void wait_async0() {
    asm volatile("s_wait_asynccnt 0" ::: "memory");
}

// ---------------------------------------------------------------- cast f32 -> bf16
__global__ __launch_bounds__(256) void cast_f32_to_bf16(
    const float* __restrict__ in, unsigned short* __restrict__ out, long n)
{
    long i = ((long)blockIdx.x * blockDim.x + threadIdx.x) * 4;
    long stride = (long)gridDim.x * blockDim.x * 4;
    for (; i < n; i += stride) {
        float4 f = *(const float4*)(in + i);
        uint2 u;
        u.x = pack2bf(f.x, f.y);
        u.y = pack2bf(f.z, f.w);
        *(uint2*)(out + i) = u;
    }
}

// ---------------------------------------------------------------- generic bf16 WMMA GEMM
// C[M x N] (f32, ldc==N) = A[M x K] (bf16 bits, row-major) * B[N x K]^T (row-major)
// block tile: 128(M) x 64(N), 8 waves, each wave 16(M) x 64(N), K-step 32.
// Double-buffered LDS fed by GLOBAL_LOAD_ASYNC_TO_LDS_B128; one barrier/iter.
__global__ __launch_bounds__(256) void gemm_bf16_wmma(
    const unsigned short* __restrict__ A,
    const unsigned short* __restrict__ B,
    float* __restrict__ C,
    int M, int N, int K)
{
    __shared__ unsigned short As[2][128 * 32];
    __shared__ unsigned short Bs[2][64 * 32];

    const int tid  = threadIdx.x;
    const int wave = tid >> 5;
    const int lane = tid & 31;
    const int l16  = lane & 15;
    const int half = lane >> 4;
    const int m0 = blockIdx.y * 128;
    const int n0 = blockIdx.x * 64;

    // per-thread staging chunks (16B each)
    const int a0r = tid >> 2,         a0c = (tid & 3) * 8;           // A chunk 0
    const int a1r = (tid + 256) >> 2, a1c = ((tid + 256) & 3) * 8;   // A chunk 1
    const int br  = tid >> 2,         bc  = (tid & 3) * 8;           // B chunk

    v8f acc[4];
#pragma unroll
    for (int s = 0; s < 4; s++) acc[s] = vzero8();

    // prologue: async-stage tile 0
    async_ld16(&As[0][a0r * 32 + a0c], &A[(size_t)(m0 + a0r) * K + a0c]);
    async_ld16(&As[0][a1r * 32 + a1c], &A[(size_t)(m0 + a1r) * K + a1c]);
    async_ld16(&Bs[0][br  * 32 + bc ], &B[(size_t)(n0 + br ) * K + bc ]);
    wait_async0();
    __syncthreads();

    int cur = 0;
    for (int k0 = 0; k0 < K; k0 += 32) {
        // prefetch next tile into the other buffer
        if (k0 + 32 < K) {
            int nxt = cur ^ 1, kn = k0 + 32;
            async_ld16(&As[nxt][a0r * 32 + a0c], &A[(size_t)(m0 + a0r) * K + kn + a0c]);
            async_ld16(&As[nxt][a1r * 32 + a1c], &A[(size_t)(m0 + a1r) * K + kn + a1c]);
            async_ld16(&Bs[nxt][br  * 32 + bc ], &B[(size_t)(n0 + br ) * K + kn + bc ]);
        }

        // A fragment (16x32), documented 16-bit A layout
        v8u au;
#pragma unroll
        for (int j = 0; j < 8; j++) {
            int k = ((j >> 2) << 4) + (half << 3) + ((j & 3) << 1);
            au[j] = *(const unsigned int*)&As[cur][(wave * 16 + l16) * 32 + k];
        }
        v16bf af = as_bf(au);

        // all 4 B fragments first, then back-to-back WMMAs
        v8u bu[4];
#pragma unroll
        for (int s = 0; s < 4; s++) {
#pragma unroll
            for (int j = 0; j < 8; j++) {
                int k = ((j >> 2) << 4) + (half << 3) + ((j & 3) << 1);
                bu[s][j] = *(const unsigned int*)&Bs[cur][(s * 16 + l16) * 32 + k];
            }
        }
#pragma unroll
        for (int s = 0; s < 4; s++) {
            acc[s] = __builtin_amdgcn_wmma_f32_16x16x32_bf16(
                false, af, false, as_bf(bu[s]), (short)0, acc[s], false, false);
        }

        wait_async0();       // next tile landed in LDS
        __syncthreads();     // everyone done reading cur / next tile visible
        cur ^= 1;
    }

#pragma unroll
    for (int s = 0; s < 4; s++) {
#pragma unroll
        for (int i = 0; i < 8; i++) {
            int m = m0 + wave * 16 + half * 8 + i;
            C[(size_t)m * N + n0 + s * 16 + l16] = acc[s][i];
        }
    }
}

// ---------------------------------------------------------------- RoPE + split + bf16
// raw: [S=1024][6144] f32 = [q(32x128) | k(8x128) | v(8x128)] per row
// out[j]    = x[2j]*cos - x[2j+1]*sin      (j in [0,64))
// out[64+j] = x[2j]*sin + x[2j+1]*cos
__global__ __launch_bounds__(64) void rope_split_kernel(
    const float* __restrict__ raw,
    const float* __restrict__ cs,   // [S][64]
    const float* __restrict__ sn,   // [S][64]
    unsigned short* __restrict__ qb,  // [32][S][128]
    unsigned short* __restrict__ kb,  // [8][S][128]
    unsigned short* __restrict__ vb)  // [8][S][128]
{
    const int s = blockIdx.x;
    const int u = blockIdx.y;
    const int j = threadIdx.x;
    if (u < 32) {
        const float* r = raw + (size_t)s * 6144 + u * 128;
        float xr = r[2 * j], xi = r[2 * j + 1];
        float c = cs[s * 64 + j], si = sn[s * 64 + j];
        unsigned short* o = qb + ((size_t)u * 1024 + s) * 128;
        o[j]      = f2bf(xr * c - xi * si);
        o[64 + j] = f2bf(xr * si + xi * c);
    } else if (u < 40) {
        int g = u - 32;
        const float* r = raw + (size_t)s * 6144 + 4096 + g * 128;
        float xr = r[2 * j], xi = r[2 * j + 1];
        float c = cs[s * 64 + j], si = sn[s * 64 + j];
        unsigned short* o = kb + ((size_t)g * 1024 + s) * 128;
        o[j]      = f2bf(xr * c - xi * si);
        o[64 + j] = f2bf(xr * si + xi * c);
    } else {
        int g = u - 40;
        const float* r = raw + (size_t)s * 6144 + 5120 + g * 128;
        unsigned short* o = vb + ((size_t)g * 1024 + s) * 128;
        o[j]      = f2bf(r[j]);
        o[64 + j] = f2bf(r[64 + j]);
    }
}

// ---------------------------------------------------------------- attention
// Per block: one q head h (g = h/4), 128 query rows. Each wave owns 16 rows.
// Reference semantics reproduced: scores = qk/sqrt(128) + (t<=s ? 1 : 0),
// softmax over MAXSEQ=2048 where positions [1024,2048) have score exactly 0
// and zero V -> denominator gets +1024*exp(-m).
__global__ __launch_bounds__(256) void attn_kernel(
    const unsigned short* __restrict__ qb,  // [32][1024][128]
    const unsigned short* __restrict__ kb,  // [8][1024][128]
    const unsigned short* __restrict__ vb,  // [8][1024][128]
    unsigned short* __restrict__ yb)        // [1024][4096]
{
    __shared__ unsigned short Ks[32 * 128];    // [t_local][e]
    __shared__ unsigned short Vst[128 * 32];   // [e][t_local]  (transposed)
    __shared__ unsigned short Ps[8 * 16 * 32]; // per-wave P staging [m][t_local]

    const int tid  = threadIdx.x;
    const int wave = tid >> 5;
    const int lane = tid & 31;
    const int l16  = lane & 15;
    const int half = lane >> 4;
    const int h = blockIdx.y;
    const int g = h >> 2;
    const int srow0 = blockIdx.x * 128 + wave * 16;
    const float scale = 0.08838834764831845f;  // 1/sqrt(128)

    // Q fragments (A layout), 4 K-chunks of 32 over HD=128
    v16bf qa[4];
    {
        const unsigned short* qrow = qb + ((size_t)h * 1024 + srow0 + l16) * 128;
#pragma unroll
        for (int kk = 0; kk < 4; kk++) {
            v8u u;
#pragma unroll
            for (int j = 0; j < 8; j++) {
                int k = kk * 32 + ((j >> 2) << 4) + (half << 3) + ((j & 3) << 1);
                u[j] = *(const unsigned int*)&qrow[k];
            }
            qa[kk] = as_bf(u);
        }
    }

    float mrow[8], lrow[8];
#pragma unroll
    for (int i = 0; i < 8; i++) { mrow[i] = -1e30f; lrow[i] = 0.0f; }
    v8f o[8];
#pragma unroll
    for (int e = 0; e < 8; e++) o[e] = vzero8();

    for (int t0 = 0; t0 < 1024; t0 += 32) {
        // K chunk: async global->LDS (row-major, no transform needed)
#pragma unroll
        for (int p = 0; p < 2; p++) {
            int c = tid + p * 256;
            int r = c >> 4, c8 = (c & 15) * 8;
            async_ld16(&Ks[r * 128 + c8],
                       &kb[((size_t)g * 1024 + t0 + r) * 128 + c8]);
        }
        // V chunk: transposed scatter into LDS (needs VGPR round-trip)
#pragma unroll
        for (int p = 0; p < 2; p++) {
            int c = tid + p * 256;
            int r = c >> 4, c8 = (c & 15) * 8;
            uint4 d = *(const uint4*)&vb[((size_t)g * 1024 + t0 + r) * 128 + c8];
            const unsigned short* e16 = (const unsigned short*)&d;
#pragma unroll
            for (int q = 0; q < 8; q++) Vst[(c8 + q) * 32 + r] = e16[q];
        }
        wait_async0();
        __syncthreads();

        // scores: 16 x 32 (two 16x16 WMMA tiles), K over HD
        v8f sc[2];
#pragma unroll
        for (int s = 0; s < 2; s++) {
            sc[s] = vzero8();
#pragma unroll
            for (int kk = 0; kk < 4; kk++) {
                v8u bu;
#pragma unroll
                for (int j = 0; j < 8; j++) {
                    int k = kk * 32 + ((j >> 2) << 4) + (half << 3) + ((j & 3) << 1);
                    bu[j] = *(const unsigned int*)&Ks[(s * 16 + l16) * 128 + k];
                }
                sc[s] = __builtin_amdgcn_wmma_f32_16x16x32_bf16(
                    false, qa[kk], false, as_bf(bu), (short)0, sc[s], false, false);
            }
        }

        // scale + additive tril bonus + online softmax (row = (half,i))
        float p0[8], p1[8];
#pragma unroll
        for (int i = 0; i < 8; i++) {
            int srow = srow0 + half * 8 + i;
            float v0 = sc[0][i] * scale + ((t0 + l16      <= srow) ? 1.0f : 0.0f);
            float v1 = sc[1][i] * scale + ((t0 + 16 + l16 <= srow) ? 1.0f : 0.0f);
            float rm = fmaxf(v0, v1);
#pragma unroll
            for (int msk = 1; msk < 16; msk <<= 1) rm = fmaxf(rm, __shfl_xor(rm, msk, 32));
            float mn = fmaxf(mrow[i], rm);
            float corr = __expf(mrow[i] - mn);
            float e0 = __expf(v0 - mn);
            float e1 = __expf(v1 - mn);
            float rs = e0 + e1;
#pragma unroll
            for (int msk = 1; msk < 16; msk <<= 1) rs += __shfl_xor(rs, msk, 32);
            lrow[i] = lrow[i] * corr + rs;
            mrow[i] = mn;
            p0[i] = e0; p1[i] = e1;
#pragma unroll
            for (int e = 0; e < 8; e++) o[e][i] *= corr;
        }

        // P: C-layout -> LDS -> A-layout (same-wave DS ops are in-order)
        {
            unsigned short* pw = &Ps[wave * 512];
#pragma unroll
            for (int i = 0; i < 8; i++) {
                pw[(half * 8 + i) * 32 + l16]      = f2bf(p0[i]);
                pw[(half * 8 + i) * 32 + 16 + l16] = f2bf(p1[i]);
            }
            v8u u;
#pragma unroll
            for (int j = 0; j < 8; j++) {
                int k = ((j >> 2) << 4) + (half << 3) + ((j & 3) << 1);
                u[j] = *(const unsigned int*)&pw[l16 * 32 + k];
            }
            v16bf pa = as_bf(u);

            // O(16x128) += P(16x32) x V(32x128): 8 WMMA over e-tiles
#pragma unroll
            for (int e = 0; e < 8; e++) {
                v8u bu;
#pragma unroll
                for (int j = 0; j < 8; j++) {
                    int k = ((j >> 2) << 4) + (half << 3) + ((j & 3) << 1);
                    bu[j] = *(const unsigned int*)&Vst[(e * 16 + l16) * 32 + k];
                }
                o[e] = __builtin_amdgcn_wmma_f32_16x16x32_bf16(
                    false, pa, false, as_bf(bu), (short)0, o[e], false, false);
            }
        }
        __syncthreads();
    }

    // finalize: include 1024 zero-score padding positions in denominator
#pragma unroll
    for (int i = 0; i < 8; i++) {
        float mf  = fmaxf(mrow[i], 0.0f);
        float s1  = __expf(mrow[i] - mf);
        float lf  = lrow[i] * s1 + 1024.0f * __expf(-mf);
        float inv = s1 / lf;
        int srow = srow0 + half * 8 + i;
        unsigned short* yr = yb + (size_t)srow * 4096 + h * 128;
#pragma unroll
        for (int e = 0; e < 8; e++)
            yr[e * 16 + l16] = f2bf(o[e][i] * inv);
    }
}

// ---------------------------------------------------------------- launch
extern "C" void kernel_launch(void* const* d_in, const int* in_sizes, int n_in,
                              void* d_out, int out_size, void* d_ws, size_t ws_size,
                              hipStream_t stream) {
    (void)in_sizes; (void)n_in; (void)out_size; (void)ws_size;
    const float* x  = (const float*)d_in[0];
    const float* fc = (const float*)d_in[1];
    const float* fs = (const float*)d_in[2];
    const float* wq = (const float*)d_in[3];
    const float* wk = (const float*)d_in[4];
    const float* wv = (const float*)d_in[5];
    const float* wo = (const float*)d_in[6];
    // d_in[7] = input_pos == arange(S): folded into the kernels analytically.

    char* ws = (char*)d_ws;
    unsigned short* xb   = (unsigned short*)(ws + 0);          //  8.39 MB
    unsigned short* wb   = (unsigned short*)(ws + 8388608);    // 50.33 MB (wq|wk|wv, 6144x4096)
    unsigned short* wob  = (unsigned short*)(ws + 58720256);   // 33.55 MB
    float*          raw  = (float*)        (ws + 92274688);    // 25.17 MB (S x 6144 f32)
    unsigned short* qbuf = (unsigned short*)(ws + 117440512);  //  8.39 MB
    unsigned short* kbuf = (unsigned short*)(ws + 125829120);  //  2.10 MB
    unsigned short* vbuf = (unsigned short*)(ws + 127926272);  //  2.10 MB
    unsigned short* ybuf = (unsigned short*)(ws + 130023424);  //  8.39 MB

    cast_f32_to_bf16<<<1024, 256, 0, stream>>>(x,  xb,             4194304L);
    cast_f32_to_bf16<<<4096, 256, 0, stream>>>(wq, wb,             16777216L);
    cast_f32_to_bf16<<<1024, 256, 0, stream>>>(wk, wb + 16777216,  4194304L);
    cast_f32_to_bf16<<<1024, 256, 0, stream>>>(wv, wb + 20971520,  4194304L);
    cast_f32_to_bf16<<<4096, 256, 0, stream>>>(wo, wob,            16777216L);

    // fused QKV projection: (1024 x 4096) * (4096 x 6144)
    gemm_bf16_wmma<<<dim3(96, 8), 256, 0, stream>>>(xb, wb, raw, 1024, 6144, 4096);
    // RoPE + split to per-head bf16 q/k/v
    rope_split_kernel<<<dim3(1024, 48), 64, 0, stream>>>(raw, fc, fs, qbuf, kbuf, vbuf);
    // attention -> y [1024 x 4096] bf16
    attn_kernel<<<dim3(8, 32), 256, 0, stream>>>(qbuf, kbuf, vbuf, ybuf);
    // output projection: (1024 x 4096) * (4096 x 4096) -> f32 d_out
    gemm_bf16_wmma<<<dim3(64, 8), 256, 0, stream>>>(ybuf, wob, (float*)d_out, 1024, 4096, 4096);
}
